// GatedGraphConvolution_40853728919776
// MI455X (gfx1250) — matrix operand order repaired and checked by reading, not compiled, not run on hardware
//
#include <hip/hip_runtime.h>

#define N_NODES 100000
#define N_EDGES 1600000
#define D 128
#define NUM_STEPS 3

typedef __attribute__((ext_vector_type(16))) __bf16        v16bf;
typedef __attribute__((ext_vector_type(8)))  float         v8f;
typedef __attribute__((ext_vector_type(4)))  unsigned int  u32x4;
typedef __attribute__((ext_vector_type(8)))  int           i32x8;
typedef __attribute__((ext_vector_type(4)))  int           i32x4;

__device__ __forceinline__ __bf16 f2bf(float f) {
  unsigned u = __builtin_bit_cast(unsigned, f);
  u += 0x7FFFu + ((u >> 16) & 1u);          // round-to-nearest-even
  unsigned short s = (unsigned short)(u >> 16);
  return __builtin_bit_cast(__bf16, s);
}

__device__ __forceinline__ float sigmoidf(float x) {
  return 1.0f / (1.0f + __expf(-x));
}

// ---------------------------------------------------------------------------
// One-shot weight conversion/transpose to bf16 so GEMM hot loops do zero cvt:
//   wt [n][k]  = weight[k][n]          (128x128)
//   cwt[n][k]  = candidate_weight[k][n](128x128)
//   ug/rg[n][k] = gate weights as-is   (128x256, already [n][k] for B-frags)
// ---------------------------------------------------------------------------
__global__ __launch_bounds__(256)
void k_prep(const float* __restrict__ weight, const float* __restrict__ cw,
            const float* __restrict__ ugW, const float* __restrict__ rgW,
            __bf16* __restrict__ wt, __bf16* __restrict__ cwt,
            __bf16* __restrict__ ug, __bf16* __restrict__ rg) {
  const int i = blockIdx.x * 256 + threadIdx.x;
  if (i < D * D) {
    const int k = i / D, n = i % D;
    wt[n * D + k]  = f2bf(weight[k * D + n]);
    cwt[n * D + k] = f2bf(cw[k * D + n]);
  }
  if (i < D * 2 * D) {
    ug[i] = f2bf(ugW[i]);
    rg[i] = f2bf(rgW[i]);
  }
}

// ---------------------------------------------------------------------------
// h = input @ W + bias ;  block = 16 rows x 8 waves; wave w -> 16-col tile
// ---------------------------------------------------------------------------
__global__ __launch_bounds__(256)
void k_input_gemm(const float* __restrict__ in, const __bf16* __restrict__ wt,
                  const float* __restrict__ bias, float* __restrict__ h) {
  __shared__ __bf16 a_s[16][D + 8];
  const int row0 = blockIdx.x * 16;
  const int tid  = threadIdx.x;
  #pragma unroll
  for (int it = 0; it < 8; ++it) {
    int j = tid + it * 256;
    int r = j >> 7, c = j & 127;
    a_s[r][c] = f2bf(in[(long)(row0 + r) * D + c]);
  }
  __syncthreads();

  const int lane = tid & 31, w = tid >> 5;
  const int hi   = lane >> 4;
  const int row  = lane & 15;
  const int n    = w * 16 + (lane & 15);

  v8f acc = {};
  #pragma unroll
  for (int kc = 0; kc < D; kc += 32) {
    const int kb = kc + hi * 16;
    v16bf a;
    #pragma unroll
    for (int i = 0; i < 16; ++i) {
      const int K = hi * 16 + i;   // A layout: contiguous k = 16*hi + i (matches B)
      a[i] = a_s[row][kc + K];
    }
    const v16bf b = *(const v16bf*)(wt + (long)n * D + kb);    // 16 contiguous bf16
    acc = __builtin_amdgcn_wmma_f32_16x16x32_bf16(false, a, false, b,
                                                  (short)0, acc, false, false);
  }
  const float bn = bias[n];
  #pragma unroll
  for (int v = 0; v < 8; ++v) {
    const int M = v + hi * 8;
    h[(long)(row0 + M) * D + n] = acc[v] + bn;
  }
}

// ---------------------------------------------------------------------------
__global__ __launch_bounds__(256)
void k_zero(float* __restrict__ p, long n) {
  long i = (long)blockIdx.x * blockDim.x + threadIdx.x;
  const long stride = (long)gridDim.x * blockDim.x;
  for (; i < n; i += stride) p[i] = 0.0f;
}

// ---------------------------------------------------------------------------
// m[erow[e]] += ew[e] * h[ecol[e]]  — 128 consecutive threads per edge:
// coalesced gather + coalesced global_atomic_add_f32 scatter (L2-resident).
// ---------------------------------------------------------------------------
__global__ __launch_bounds__(256)
void k_spmm(const int* __restrict__ erow, const int* __restrict__ ecol,
            const float* __restrict__ ew, const float* __restrict__ h,
            float* __restrict__ m) {
  const long idx = (long)blockIdx.x * 256 + threadIdx.x;
  const int e = (int)(idx >> 7);
  const int d = (int)(idx & 127);
  if (e < N_EDGES) {
    const int src = ecol[e];
    const int dst = erow[e];
    const float v = ew[e] * h[(long)src * D + d];
    atomicAdd(&m[(long)dst * D + d], v);
  }
}

// ---------------------------------------------------------------------------
// Fused GRU-style update for 16 rows:
//   g = [h | m];  z = sigmoid(g @ ugW.T + ugb);  r = sigmoid(g @ rgW.T + rgb)
//   cand = tanh((r*h) @ cw);  h' = z*h + (1-z)*cand
// f32 h tile is DMA'd to LDS by the Tensor Data Mover while the waves stage
// the bf16 [h|m] copy used by the WMMAs.
// ---------------------------------------------------------------------------
__global__ __launch_bounds__(256)
void k_gates(const float* __restrict__ h, const float* __restrict__ m,
             const __bf16* __restrict__ ug, const float* __restrict__ ugb,
             const __bf16* __restrict__ rg, const float* __restrict__ rgb,
             const __bf16* __restrict__ cwt, float* __restrict__ out) {
  __shared__ float  h_s[16][D];          // written by TDM (contiguous 16x128 f32)
  __shared__ __bf16 g_s[16][2 * D + 8];  // bf16 [h | m] for A fragments
  __shared__ __bf16 rh_s[16][D + 8];

  const int row0 = blockIdx.x * 16;
  const int tid  = threadIdx.x;
  const int lane = tid & 31, w = tid >> 5;

  // ---- wave 0: kick off TDM copy of the f32 h tile into LDS
  if (w == 0) {
    const unsigned lds = (unsigned)(unsigned long long)(const void*)&h_s[0][0];
    const unsigned long long ga =
        (unsigned long long)(const void*)(h + (long)row0 * D);
    u32x4 g0 = { 1u,                                    // count=1, no gather
                 lds,                                   // lds_addr
                 (unsigned)(ga & 0xFFFFFFFFu),          // global_addr lo
                 (unsigned)((ga >> 32) & 0x1FFFFFFu) | (2u << 30) }; // hi | type=2
    i32x8 g1 = { (int)(2u << 16),        // data_size = 4 bytes
                 (int)(128u << 16),      // tensor_dim0 = 128
                 (int)(16u << 16),       // tensor_dim1 = 16
                 (int)(128u << 16),      // tile_dim0   = 128
                 16,                     // tile_dim1   = 16
                 128,                    // tensor_dim0_stride = 128
                 0, 0 };
    i32x4 gz = { 0, 0, 0, 0 };
#if defined(__clang_major__) && (__clang_major__ >= 23)
    i32x8 gz8 = { 0, 0, 0, 0, 0, 0, 0, 0 };
    __builtin_amdgcn_tensor_load_to_lds(g0, g1, gz, gz, gz8, 0);
#else
    __builtin_amdgcn_tensor_load_to_lds(g0, g1, gz, gz, 0);
#endif
  }

  // ---- all waves: stage bf16 g = [h | m] straight from global (L2-resident)
  #pragma unroll
  for (int it = 0; it < 8; ++it) {
    int j = tid + it * 256;
    int r = j >> 7, c = j & 127;
    g_s[r][c]     = f2bf(h[(long)(row0 + r) * D + c]);
    g_s[r][D + c] = f2bf(m[(long)(row0 + r) * D + c]);
  }
  if (w == 0) __builtin_amdgcn_s_wait_tensorcnt(0);
  __syncthreads();

  const int hi  = lane >> 4;
  const int row = lane & 15;
  const int n   = w * 16 + (lane & 15);

  // ---- z / r gates: K = 256 over g; one A fragment feeds both WMMAs
  v8f accz = {}, accr = {};
  #pragma unroll
  for (int kc = 0; kc < 2 * D; kc += 32) {
    const int kb = kc + hi * 16;
    v16bf a;
    #pragma unroll
    for (int i = 0; i < 16; ++i) {
      const int K = hi * 16 + i;
      a[i] = g_s[row][kc + K];
    }
    const v16bf bz = *(const v16bf*)(ug + (long)n * (2 * D) + kb);
    const v16bf br = *(const v16bf*)(rg + (long)n * (2 * D) + kb);
    accz = __builtin_amdgcn_wmma_f32_16x16x32_bf16(false, a, false, bz,
                                                   (short)0, accz, false, false);
    accr = __builtin_amdgcn_wmma_f32_16x16x32_bf16(false, a, false, br,
                                                   (short)0, accr, false, false);
  }

  const float zb = ugb[n], rb = rgb[n];
  float zreg[8];
  #pragma unroll
  for (int v = 0; v < 8; ++v) {
    const int M = v + hi * 8;
    const float z = sigmoidf(accz[v] + zb);
    const float r = sigmoidf(accr[v] + rb);
    zreg[v] = z;
    rh_s[M][n] = f2bf(r * h_s[M][n]);
  }
  __syncthreads();

  // ---- cand = tanh((r*h) @ cw)
  v8f accc = {};
  #pragma unroll
  for (int kc = 0; kc < D; kc += 32) {
    const int kb = kc + hi * 16;
    v16bf a;
    #pragma unroll
    for (int i = 0; i < 16; ++i) {
      const int K = hi * 16 + i;
      a[i] = rh_s[row][kc + K];
    }
    const v16bf b = *(const v16bf*)(cwt + (long)n * D + kb);
    accc = __builtin_amdgcn_wmma_f32_16x16x32_bf16(false, a, false, b,
                                                   (short)0, accc, false, false);
  }

  #pragma unroll
  for (int v = 0; v < 8; ++v) {
    const int M = v + hi * 8;
    const float cand = tanhf(accc[v]);
    const float hv   = h_s[M][n];
    out[(long)(row0 + M) * D + n] = zreg[v] * hv + (1.0f - zreg[v]) * cand;
  }
}

// ---------------------------------------------------------------------------
extern "C" void kernel_launch(void* const* d_in, const int* in_sizes, int n_in,
                              void* d_out, int out_size, void* d_ws, size_t ws_size,
                              hipStream_t stream) {
  const float* input  = (const float*)d_in[0];
  const int*   erow   = (const int*)  d_in[1];
  const int*   ecol   = (const int*)  d_in[2];
  const float* ew     = (const float*)d_in[3];
  const float* weight = (const float*)d_in[4];
  const float* bias   = (const float*)d_in[5];
  const float* cw     = (const float*)d_in[6];
  const float* ugW    = (const float*)d_in[7];
  const float* ugb    = (const float*)d_in[8];
  const float* rgW    = (const float*)d_in[9];
  const float* rgb    = (const float*)d_in[10];
  float* out = (float*)d_out;

  // workspace layout
  float*  h   = (float*)d_ws;                          // N*D f32
  float*  m   = h + (size_t)N_NODES * D;               // N*D f32
  __bf16* wt  = (__bf16*)(m + (size_t)N_NODES * D);    // 128*128
  __bf16* cwt = wt + D * D;                            // 128*128
  __bf16* ug  = cwt + D * D;                           // 128*256
  __bf16* rg  = ug + D * 2 * D;                        // 128*256

  const int tiles = N_NODES / 16;                      // 6250, exact

  k_prep<<<(D * 2 * D + 255) / 256, 256, 0, stream>>>(weight, cw, ugW, rgW,
                                                      wt, cwt, ug, rg);
  k_input_gemm<<<tiles, 256, 0, stream>>>(input, wt, bias, h);

  for (int s = 0; s < NUM_STEPS; ++s) {
    k_zero<<<2048, 256, 0, stream>>>(m, (long)N_NODES * D);
    k_spmm<<<(int)(((long)N_EDGES * D) / 256), 256, 0, stream>>>(erow, ecol, ew, h, m);
    float* dst = (s == NUM_STEPS - 1) ? out : h;       // last step -> d_out
    k_gates<<<tiles, 256, 0, stream>>>(h, m, ug, ugb, rg, rgb, cwt, dst);
  }
}
